// MambaUV_73796128080205
// MI455X (gfx1250) — matrix operand order repaired and checked by reading, not compiled, not run on hardware
//
#include <hip/hip_runtime.h>

// ---------------------------------------------------------------------------
// Mamba stack forward for MI455X (gfx1250, wave32, WMMA).
// B=16, C_IN=3, L=1024, D_MODEL=256, N_LAYERS=4, D_STATE=16, D_CONV=4,
// D_INNER=512, DT_RANK=16.  GEMMs via v_wmma_f32_16x16x32_bf16 with
// 16x(16*NT) wave tiles and a manual 1-deep software pipeline over K-chunks
// (next chunk's loads issued before current chunk's WMMAs -> partial
// s_wait_loadcnt, in-wave load/compute overlap).
// ---------------------------------------------------------------------------

#define B_SZ   16
#define L_SZ   1024
#define DM     256          // d_model
#define DI     512          // d_inner
#define DS     16           // d_state
#define DTR    16           // dt_rank
#define NL     4            // layers
#define M_TOK  (B_SZ * L_SZ)   // 16384 tokens

typedef __attribute__((ext_vector_type(16))) __bf16 v16bf;
typedef __attribute__((ext_vector_type(8)))  float  v8f;
typedef __attribute__((ext_vector_type(4)))  unsigned int u32x4;

union FragU { u32x4 q[2]; v16bf v; };   // 32 bytes = 16 bf16 = one WMMA operand

// ------------------------------ helpers ------------------------------------

__device__ __forceinline__ float wave_sum(float v) {
#pragma unroll
  for (int m = 16; m >= 1; m >>= 1) v += __shfl_xor(v, m, 32);
  return v;
}

__device__ __forceinline__ float silu_f(float x) {
  return x / (1.f + __expf(-x));
}

// ------------------------- fp32 -> bf16 convert ----------------------------

__global__ __launch_bounds__(256)
void k_convert_bf16(const float* __restrict__ src, __bf16* __restrict__ dst, int n) {
  int i = blockIdx.x * 256 + threadIdx.x;
  if (i < n) dst[i] = (__bf16)src[i];
}

// ------------------------------ stem conv ----------------------------------
// h[b,t,dm] = relu( sum_{c,j} x[b,c,t-1+j] * w[dm,c,j] + bias[dm] )
__global__ __launch_bounds__(256)
void k_stem(const float* __restrict__ x, const float* __restrict__ w,
            const float* __restrict__ bias, float* __restrict__ h) {
  const int bt = blockIdx.x;
  const int b  = bt >> 10;
  const int t  = bt & (L_SZ - 1);
  const int dm = threadIdx.x;
  float acc = bias[dm];
#pragma unroll
  for (int c = 0; c < 3; ++c) {
    const float* xc = x + (size_t)(b * 3 + c) * L_SZ;
    const float* wc = w + (dm * 3 + c) * 3;
#pragma unroll
    for (int j = 0; j < 3; ++j) {
      const int tt = t - 1 + j;
      if (tt >= 0 && tt < L_SZ) acc += xc[tt] * wc[j];
    }
  }
  h[(size_t)bt * DM + dm] = fmaxf(acc, 0.f);
}

// ------------------------ layernorm -> bf16 (GEMM A) -----------------------
// One wave per token row of 256; lane handles 8 consecutive channels.
__global__ __launch_bounds__(256)
void k_layernorm_bf16(const float* __restrict__ h, const float* __restrict__ g,
                      const float* __restrict__ be, __bf16* __restrict__ out) {
  const int row  = blockIdx.x * 8 + (threadIdx.x >> 5);
  const int lane = threadIdx.x & 31;
  const int c0   = lane * 8;
  const float* p = h + (size_t)row * DM + c0;
  float v[8];
  float s = 0.f;
#pragma unroll
  for (int i = 0; i < 8; ++i) { v[i] = p[i]; s += v[i]; }
  const float mu = wave_sum(s) * (1.f / DM);
  float q = 0.f;
#pragma unroll
  for (int i = 0; i < 8; ++i) { const float d = v[i] - mu; q += d * d; }
  const float rs = rsqrtf(wave_sum(q) * (1.f / DM) + 1e-5f);
  __bf16* o = out + (size_t)row * DM + c0;
#pragma unroll
  for (int i = 0; i < 8; ++i)
    o[i] = (__bf16)((v[i] - mu) * rs * g[c0 + i] + be[c0 + i]);
}

// ----------------------------- WMMA GEMM -----------------------------------
// C[M,N] = A[M,K] * W[N,K]^T (+ optional residual Cin), bf16 in / f32 out.
// 128 threads = 4 waves; wave owns a 16x(16*NT) tile (A fragment reused
// across NT column tiles).  K is a compile-time constant; the K-chunk loop
// is fully unrolled with a 1-deep software pipeline: loads for chunk k+32
// are issued before the WMMAs of chunk k, so the WMMAs consume registers
// loaded one stage earlier and the loads get a full stage of latency cover.
// Fragment layouts per CDNA5 ISA 7.12.2 (wave32):
//   A 16x32 : lanes 0-15 row M=lane hold K {k..k+7, k+16..k+23};
//             lanes 16-31 same rows hold K {k+8..k+15, k+24..k+31}.
//   B 32x16 : lane n holds column N=n; lanes 0-15 K {k..k+7, k+16..k+23},
//             lanes 16-31 K {k+8..k+15, k+24..k+31} (same as A).
//   D       : lane l VGPR r -> C[m0 + 8*(l>>4) + r, n0 + (l&15)].
template <int NT, bool HASC, int KC>
__global__ __launch_bounds__(128)
void k_gemm_bf16(const __bf16* __restrict__ A, const __bf16* __restrict__ W,
                 const float* __restrict__ Cin, float* __restrict__ Cout,
                 int M, int N) {
  const int wave = threadIdx.x >> 5;
  const int lane = threadIdx.x & 31;
  const int row  = lane & 15;
  const int hi   = lane >> 4;
  const int m0 = (blockIdx.x * 4 + wave) * 16;
  const int n0 = blockIdx.y * (16 * NT);

  const __bf16* ap = A + (size_t)(m0 + row) * KC + hi * 8;
  const __bf16* wp = W + (size_t)(n0 + row) * KC + hi * 8;
  const size_t wstride = (size_t)16 * KC;   // next 16-column tile of W

  v8f acc[NT];
#pragma unroll
  for (int j = 0; j < NT; ++j) acc[j] = (v8f){};

  FragU fa, fb[NT];
  // ---- prologue: load chunk 0 ----
  fa.q[0] = *(const u32x4*)(ap);
  fa.q[1] = *(const u32x4*)(ap + 16);
#pragma unroll
  for (int j = 0; j < NT; ++j) {
    const __bf16* wj = wp + (size_t)j * wstride;
    fb[j].q[0] = *(const u32x4*)(wj);
    fb[j].q[1] = *(const u32x4*)(wj + 16);
  }

  // ---- pipelined main loop (fully unrolled, copies absorbed by SSA) ----
#pragma unroll
  for (int k = 32; k < KC; k += 32) {
    FragU na, nb[NT];
    na.q[0] = *(const u32x4*)(ap + k);
    na.q[1] = *(const u32x4*)(ap + k + 16);
#pragma unroll
    for (int j = 0; j < NT; ++j) {
      const __bf16* wj = wp + (size_t)j * wstride;
      nb[j].q[0] = *(const u32x4*)(wj + k);
      nb[j].q[1] = *(const u32x4*)(wj + k + 16);
    }
#pragma unroll
    for (int j = 0; j < NT; ++j) {
      acc[j] = __builtin_amdgcn_wmma_f32_16x16x32_bf16(
          false, fa.v, false, fb[j].v, (short)0, acc[j], false, false);
    }
    fa = na;
#pragma unroll
    for (int j = 0; j < NT; ++j) fb[j] = nb[j];
  }

  // ---- epilogue: last chunk ----
#pragma unroll
  for (int j = 0; j < NT; ++j) {
    acc[j] = __builtin_amdgcn_wmma_f32_16x16x32_bf16(
        false, fa.v, false, fb[j].v, (short)0, acc[j], false, false);
  }

  const int r0 = m0 + hi * 8;
#pragma unroll
  for (int j = 0; j < NT; ++j) {
    const int col = n0 + j * 16 + row;
#pragma unroll
    for (int r = 0; r < 8; ++r) {
      const size_t idx = (size_t)(r0 + r) * N + col;
      float v = acc[j][r];
      if (HASC) v += Cin[idx];
      Cout[idx] = v;
    }
  }
}

// --------------------- causal depthwise conv + SiLU ------------------------
// xz layout (B*L, 2*DI); xin = channels [0,DI).  Writes f32 and bf16 copies.
__global__ __launch_bounds__(256)
void k_dwconv_silu(const float* __restrict__ xz, const float* __restrict__ cw,
                   const float* __restrict__ cb, float* __restrict__ xs,
                   __bf16* __restrict__ xsb) {
  const int gid  = blockIdx.x * 256 + threadIdx.x;
  const int d    = gid & (DI - 1);
  const int bt   = gid >> 9;
  const int t    = bt & (L_SZ - 1);
  const int base = bt - t;   // b*L
  float acc = cb[d];
#pragma unroll
  for (int j = 0; j < 4; ++j) {
    const int tt = t - 3 + j;
    if (tt >= 0) acc += xz[(size_t)(base + tt) * (2 * DI) + d] * cw[d * 4 + j];
  }
  acc = silu_f(acc);
  xs[(size_t)bt * DI + d]  = acc;
  xsb[(size_t)bt * DI + d] = (__bf16)acc;
}

// --------------------------- selective scan --------------------------------
// One thread per (b, d); 16 states in registers; dt-proj + softplus + SiLU
// gating fused.  proj rows (48 floats/token) staged through LDS per 16 steps.
__global__ __launch_bounds__(256)
void k_scan(const float* __restrict__ xs, const float* __restrict__ xz,
            const float* __restrict__ proj, const float* __restrict__ dtw,
            const float* __restrict__ dtb, const float* __restrict__ alog,
            const float* __restrict__ Dp, __bf16* __restrict__ y) {
  const int b = blockIdx.x >> 1;
  const int d = ((blockIdx.x & 1) << 8) + threadIdx.x;

  float a[DS], w[DTR], h[DS];
#pragma unroll
  for (int s = 0; s < DS; ++s) { a[s] = -__expf(alog[d * DS + s]); h[s] = 0.f; }
#pragma unroll
  for (int r = 0; r < DTR; ++r) w[r] = dtw[d * DTR + r];
  const float bias = dtb[d];
  const float Dd   = Dp[d];

  __shared__ float sh[16 * 48];
  const int bbase = b << 10;

  for (int t0 = 0; t0 < L_SZ; t0 += 16) {
    __syncthreads();
    for (int i = threadIdx.x; i < 16 * 48; i += 256)
      sh[i] = proj[(size_t)(bbase + t0) * 48 + i];
    __syncthreads();
#pragma unroll 2
    for (int tt = 0; tt < 16; ++tt) {
      const float* pr = sh + tt * 48;
      float dtv = bias;
#pragma unroll
      for (int r = 0; r < DTR; ++r) dtv += w[r] * pr[r];
      const float delta = (dtv > 20.f) ? dtv : log1pf(__expf(dtv));  // softplus
      const int t = t0 + tt;
      const float u  = xs[(size_t)(bbase + t) * DI + d];
      const float du = delta * u;
      float yv = 0.f;
#pragma unroll
      for (int s = 0; s < DS; ++s) {
        h[s] = h[s] * __expf(delta * a[s]) + du * pr[DTR + s];
        yv  += h[s] * pr[DTR + DS + s];
      }
      yv += u * Dd;
      const float z = xz[(size_t)(bbase + t) * (2 * DI) + DI + d];
      yv *= silu_f(z);
      y[(size_t)(bbase + t) * DI + d] = (__bf16)yv;
    }
  }
}

// ----------------------- final LN + head + sigmoid -------------------------
__global__ __launch_bounds__(256)
void k_head(const float* __restrict__ h, const float* __restrict__ g,
            const float* __restrict__ be, const float* __restrict__ hw,
            const float* __restrict__ hb, float* __restrict__ out) {
  const int row  = blockIdx.x * 8 + (threadIdx.x >> 5);
  const int lane = threadIdx.x & 31;
  const int c0   = lane * 8;
  const float* p = h + (size_t)row * DM + c0;
  float v[8];
  float s = 0.f;
#pragma unroll
  for (int i = 0; i < 8; ++i) { v[i] = p[i]; s += v[i]; }
  const float mu = wave_sum(s) * (1.f / DM);
  float q = 0.f;
#pragma unroll
  for (int i = 0; i < 8; ++i) { const float d = v[i] - mu; q += d * d; }
  const float rs = rsqrtf(wave_sum(q) * (1.f / DM) + 1e-5f);
  float acc = 0.f;
#pragma unroll
  for (int i = 0; i < 8; ++i)
    acc += ((v[i] - mu) * rs * g[c0 + i] + be[c0 + i]) * hw[c0 + i];
  const float tot = wave_sum(acc);
  if (lane == 0) out[row] = 1.f / (1.f + __expf(-(tot + hb[0])));
}

// ------------------------------- launcher ----------------------------------

extern "C" void kernel_launch(void* const* d_in, const int* in_sizes, int n_in,
                              void* d_out, int out_size, void* d_ws, size_t ws_size,
                              hipStream_t stream) {
  const float* x       = (const float*)d_in[0];
  const float* stem_w  = (const float*)d_in[1];
  const float* stem_b  = (const float*)d_in[2];
  const float* norm_g  = (const float*)d_in[3];
  const float* norm_b  = (const float*)d_in[4];
  const float* in_w    = (const float*)d_in[5];   // (NL, 2*DI, DM)
  const float* conv_w  = (const float*)d_in[6];   // (NL, DI, 4)
  const float* conv_b  = (const float*)d_in[7];   // (NL, DI)
  const float* xproj_w = (const float*)d_in[8];   // (NL, 48, DI)
  const float* dt_w    = (const float*)d_in[9];   // (NL, DI, DTR)
  const float* dt_b    = (const float*)d_in[10];  // (NL, DI)
  const float* A_log   = (const float*)d_in[11];  // (NL, DI, DS)
  const float* Dp      = (const float*)d_in[12];  // (NL, DI)
  const float* out_w   = (const float*)d_in[13];  // (NL, DM, DI)
  const float* fnorm_g = (const float*)d_in[14];
  const float* fnorm_b = (const float*)d_in[15];
  const float* head_w  = (const float*)d_in[16];
  const float* head_b  = (const float*)d_in[17];
  (void)in_sizes; (void)n_in; (void)out_size; (void)ws_size;

  // ---- workspace carve-out (256B aligned), ~158 MB total ----
  char* ws = (char*)d_ws;
  size_t off = 0;
  auto carve = [&](size_t bytes) -> char* {
    char* p = ws + off;
    off += (bytes + 255) & ~(size_t)255;
    return p;
  };
  __bf16* wbf_in = (__bf16*)carve((size_t)NL * 2 * DI * DM * sizeof(__bf16));
  __bf16* wbf_x  = (__bf16*)carve((size_t)NL * 48 * DI     * sizeof(__bf16));
  __bf16* wbf_o  = (__bf16*)carve((size_t)NL * DM * DI     * sizeof(__bf16));
  float*  hres   = (float*) carve((size_t)M_TOK * DM       * sizeof(float));
  __bf16* xn     = (__bf16*)carve((size_t)M_TOK * DM       * sizeof(__bf16));
  float*  xz     = (float*) carve((size_t)M_TOK * 2 * DI   * sizeof(float));
  float*  xs     = (float*) carve((size_t)M_TOK * DI       * sizeof(float));
  __bf16* xsb    = (__bf16*)carve((size_t)M_TOK * DI       * sizeof(__bf16));
  float*  proj   = (float*) carve((size_t)M_TOK * 48       * sizeof(float));
  __bf16* yb     = (__bf16*)carve((size_t)M_TOK * DI       * sizeof(__bf16));

  // ---- weight precision conversion (once per launch, deterministic) ----
  {
    const int n1 = NL * 2 * DI * DM;
    k_convert_bf16<<<(n1 + 255) / 256, 256, 0, stream>>>(in_w, wbf_in, n1);
    const int n2 = NL * 48 * DI;
    k_convert_bf16<<<(n2 + 255) / 256, 256, 0, stream>>>(xproj_w, wbf_x, n2);
    const int n3 = NL * DM * DI;
    k_convert_bf16<<<(n3 + 255) / 256, 256, 0, stream>>>(out_w, wbf_o, n3);
  }

  // ---- stem ----
  k_stem<<<M_TOK, 256, 0, stream>>>(x, stem_w, stem_b, hres);

  // ---- layers ----
  for (int i = 0; i < NL; ++i) {
    k_layernorm_bf16<<<M_TOK / 8, 256, 0, stream>>>(
        hres, norm_g + i * DM, norm_b + i * DM, xn);

    // in_proj: M=16384, N=1024, K=256; wave tile 16x64 (NT=4)
    dim3 g1(M_TOK / 64, (2 * DI) / 64);
    k_gemm_bf16<4, false, DM><<<g1, 128, 0, stream>>>(
        xn, wbf_in + (size_t)i * 2 * DI * DM, nullptr, xz, M_TOK, 2 * DI);

    k_dwconv_silu<<<(M_TOK * DI) / 256, 256, 0, stream>>>(
        xz, conv_w + i * DI * 4, conv_b + i * DI, xs, xsb);

    // x_proj: M=16384, N=48, K=512; wave tile 16x48 (NT=3)
    dim3 g2(M_TOK / 64, 1);
    k_gemm_bf16<3, false, DI><<<g2, 128, 0, stream>>>(
        xsb, wbf_x + (size_t)i * 48 * DI, nullptr, proj, M_TOK, 48);

    k_scan<<<B_SZ * (DI / 256), 256, 0, stream>>>(
        xs, xz, proj, dt_w + i * DI * DTR, dt_b + i * DI,
        A_log + i * DI * DS, Dp + i * DI, yb);

    // out_proj + residual: M=16384, N=256, K=512; wave tile 16x64 (NT=4)
    dim3 g3(M_TOK / 64, DM / 64);
    k_gemm_bf16<4, true, DI><<<g3, 128, 0, stream>>>(
        yb, wbf_o + (size_t)i * DM * DI, hres, hres, M_TOK, DM);
  }

  // ---- final LN + head + sigmoid ----
  k_head<<<M_TOK / 8, 256, 0, stream>>>(
      hres, fnorm_g, fnorm_b, head_w, head_b, (float*)d_out);
}
